// BiLSTM_CRF_22376779612729
// MI455X (gfx1250) — matrix operand order repaired and checked by reading, not compiled, not run on hardware
//
#include <hip/hip_runtime.h>
#include <hip/hip_bf16.h>
#include <stdint.h>

typedef __attribute__((ext_vector_type(16))) _Float16 v16h;
typedef __attribute__((ext_vector_type(8)))  float    v8f;

#define Tlen 256
#define Bb   64
#define Ee   256
#define Hh   256
#define G4   1024   // 4*H
#define Kk   9
#define KDIM 512    // E + H

// ---------------- prep kernels ----------------

__global__ void embed_gather_f16(const int* __restrict__ sent,
                                 const float* __restrict__ emb,
                                 _Float16* __restrict__ x16) {
  int row = blockIdx.x;             // row = t*B + b  (t-major)
  int t = row >> 6, b = row & 63;
  int tok = sent[(size_t)b * Tlen + t];
  int e = threadIdx.x;              // E == 256 threads
  x16[(size_t)row * Ee + e] = (_Float16)emb[(size_t)tok * Ee + e];
}

// WT[k][n] = (k < E) ? w_ih[n][k] : w_hh[n][k-E]   (k-major so WMMA B frags are contiguous)
__global__ void build_wt(const float* __restrict__ w_ih_f, const float* __restrict__ w_hh_f,
                         const float* __restrict__ w_ih_b, const float* __restrict__ w_hh_b,
                         _Float16* __restrict__ WTf, _Float16* __restrict__ WTb) {
  int k   = blockIdx.x;             // 0..511
  int dir = blockIdx.y;             // 0 fwd, 1 bwd
  const float* wi = dir ? w_ih_b : w_ih_f;
  const float* wh = dir ? w_hh_b : w_hh_f;
  _Float16* WT = dir ? WTb : WTf;
  for (int n = threadIdx.x; n < G4; n += blockDim.x) {
    float v = (k < Ee) ? wi[(size_t)n * Ee + k] : wh[(size_t)n * Hh + (k - Ee)];
    WT[(size_t)k * G4 + n] = (_Float16)v;
  }
}

__global__ void build_bsum(const float* __restrict__ bihf, const float* __restrict__ bhhf,
                           const float* __restrict__ bihb, const float* __restrict__ bhhb,
                           float* __restrict__ bsf, float* __restrict__ bsb) {
  int n = blockIdx.x * blockDim.x + threadIdx.x;
  if (n < G4) { bsf[n] = bihf[n] + bhhf[n]; bsb[n] = bihb[n] + bhhb[n]; }
}

// ---------------- WMMA fragment loaders (CDNA5 16-bit layouts, wave32) ----------------

// A 16x32 f16: lane<->M (row = lane&15), VGPR<->K. VGPR v holds packed pair
// k0 = (v/4)*16 + (lane>=16)*8 + (v%4)*2  (pairs contiguous -> b32 LDS reads)
__device__ __forceinline__ v16h load_a_frag(const _Float16* Ash, int mt, int kb, int lane) {
  int row  = mt * 16 + (lane & 15);
  int half = lane >> 4;
  const uint32_t* p = (const uint32_t*)(Ash + row * KDIM);
  union { v16h v; uint32_t u[8]; } r;
#pragma unroll
  for (int v = 0; v < 8; ++v) {
    int k0 = ((v >> 2) << 4) + (half << 3) + ((v & 3) << 1);
    r.u[v] = p[(kb + k0) >> 1];
  }
  return r.v;
}

// B 32x16 f16: lane<->N (n = nb + lane&15), element s<->K in the same order as A:
// k = (s/8)*16 + (lane>=16)*8 + (s%8)  (strided gather from k-major WT).
__device__ __forceinline__ v16h load_b_frag(const _Float16* WT, int nb, int kb, int lane) {
  const _Float16* p = WT + (size_t)kb * G4 + nb + (lane & 15);
  int half = lane >> 4;
  v16h r;
#pragma unroll
  for (int s = 0; s < 16; ++s) {
    int k = ((s >> 3) << 4) + (half << 3) + (s & 7);
    r[s] = p[(size_t)k * G4];
  }
  return r;
}

__device__ __forceinline__ float sigf(float x) { return 1.0f / (1.0f + expf(-x)); }

// ---------------- persistent BiLSTM scan: block 0 = forward, block 1 = backward ----------------

__global__ __launch_bounds__(1024) void lstm_scan(
    const _Float16* __restrict__ x16,
    const _Float16* __restrict__ WTf, const _Float16* __restrict__ WTb,
    const float* __restrict__ bsf, const float* __restrict__ bsb,
    _Float16* __restrict__ hsf, _Float16* __restrict__ hsb) {
  extern __shared__ char smem[];
  _Float16* Ash = (_Float16*)smem;                                   // [64][512] f16  (x | h)
  _Float16* Gsh = (_Float16*)(smem + (size_t)Bb * KDIM * 2);         // [64][1024] f16 gates
  float*    Csh = (float*)(smem + (size_t)Bb * KDIM * 2 + (size_t)Bb * G4 * 2); // [64][256] f32 cell

  const int dir = blockIdx.x;
  const _Float16* WT = dir ? WTb : WTf;
  const float* bsum  = dir ? bsb : bsf;
  _Float16* hs       = dir ? hsb : hsf;

  const int tid  = threadIdx.x;
  const int lane = tid & 31;
  const int wv   = tid >> 5;      // 0..31
  const int mt   = wv & 3;        // M tile (batch)   : 4 tiles of 16
  const int ng   = wv >> 2;       // N tile group     : 8 groups x 8 tiles
  const int half = lane >> 4;

  // init c = 0 and h-half of A = 0
  for (int i = tid; i < Bb * Hh; i += 1024) {
    Csh[i] = 0.0f;
    int b = i >> 8, j = i & 255;
    Ash[b * KDIM + Ee + j] = (_Float16)0.0f;
  }
  __syncthreads();

  for (int s = 0; s < Tlen; ++s) {
    int t = dir ? (Tlen - 1 - s) : s;

    // stage x_t into A[:, 0:E]  (each thread copies 16 f16 = 32B)
    {
      int b = tid >> 4;
      int e = (tid & 15) << 4;
      const uint4* src = (const uint4*)(x16 + ((size_t)t * Bb + b) * Ee + e);
      uint4* dst = (uint4*)(Ash + b * KDIM + e);
      dst[0] = src[0];
      dst[1] = src[1];
      if (s + 1 < Tlen) {
        int tn = dir ? (Tlen - 2 - s) : (s + 1);
        __builtin_prefetch(x16 + ((size_t)tn * Bb + b) * Ee + e, 0, 1);
      }
    }
    __syncthreads();

    // gates(64x1024) = [x_t | h_{t-1}](64x512) x W^T   via v_wmma_f32_16x16x32_f16
    v8f acc[8] = {};
#pragma unroll 1
    for (int kk = 0; kk < 16; ++kk) {
      int kb = kk * 32;
      v16h a = load_a_frag(Ash, mt, kb, lane);
#pragma unroll
      for (int i = 0; i < 8; ++i) {
        int nb = (ng * 8 + i) * 16;
        v16h bf = load_b_frag(WT, nb, kb, lane);
        acc[i] = __builtin_amdgcn_wmma_f32_16x16x32_f16(
            false, a, false, bf, (short)0, acc[i], false, false);
      }
    }
    // C/D layout: VGPR r -> M = r + 8*half, N = lane&15
#pragma unroll
    for (int i = 0; i < 8; ++i) {
      int col = (ng * 8 + i) * 16 + (lane & 15);
#pragma unroll
      for (int r = 0; r < 8; ++r) {
        int m = mt * 16 + r + 8 * half;
        Gsh[m * G4 + col] = (_Float16)acc[i][r];
      }
    }
    __syncthreads();

    // elementwise LSTM cell (gate order i,f,g,o as in reference split)
#pragma unroll 1
    for (int rr = 0; rr < 16; ++rr) {
      int flat = tid + rr * 1024;       // 64*256 states
      int b = flat >> 8, j = flat & 255;
      float gi = (float)Gsh[b * G4 + j]       + bsum[j];
      float gf = (float)Gsh[b * G4 + j + 256] + bsum[j + 256];
      float gg = (float)Gsh[b * G4 + j + 512] + bsum[j + 512];
      float go = (float)Gsh[b * G4 + j + 768] + bsum[j + 768];
      float c = sigf(gf) * Csh[b * Hh + j] + sigf(gi) * tanhf(gg);
      float h = sigf(go) * tanhf(c);
      Csh[b * Hh + j] = c;
      _Float16 h16 = (_Float16)h;
      Ash[b * KDIM + Ee + j] = h16;                       // feed next step
      hs[((size_t)t * Bb + b) * Hh + j] = h16;            // time-aligned output
    }
    __syncthreads();
  }
}

// ---------------- emission + CRF (one block per batch element) ----------------

__global__ void emis_crf(const _Float16* __restrict__ hsf, const _Float16* __restrict__ hsb,
                         const float* __restrict__ w_out, const float* __restrict__ b_out,
                         const float* __restrict__ start_t, const float* __restrict__ end_t,
                         const float* __restrict__ trans,
                         const int* __restrict__ tags, const unsigned char* __restrict__ mask,
                         float* __restrict__ nll_b) {
  int b = blockIdx.x;
  int tid = threadIdx.x;                 // 256 threads == T
  __shared__ float wsh[Kk][512];
  __shared__ float bsh[Kk];
  __shared__ float trsh[Kk * Kk];
  __shared__ float em[Tlen][Kk];

  for (int i = tid; i < Kk * 512; i += 256) wsh[i / 512][i % 512] = w_out[i];
  if (tid < Kk) bsh[tid] = b_out[tid];
  if (tid < Kk * Kk) trsh[tid] = trans[tid];
  __syncthreads();

  // emis[t][k] = hf(t,b,:).w_out[k][:256] + hb(t,b,:).w_out[k][256:] + b_out[k]
  {
    int t = tid;
    float acc[Kk];
#pragma unroll
    for (int k = 0; k < Kk; ++k) acc[k] = bsh[k];
    const _Float16* pf = hsf + ((size_t)t * Bb + b) * Hh;
    const _Float16* pb = hsb + ((size_t)t * Bb + b) * Hh;
    for (int j = 0; j < Hh; ++j) {
      float vf = (float)pf[j];
      float vb = (float)pb[j];
#pragma unroll
      for (int k = 0; k < Kk; ++k)
        acc[k] += vf * wsh[k][j] + vb * wsh[k][Hh + j];
    }
#pragma unroll
    for (int k = 0; k < Kk; ++k) em[t][k] = acc[k];
  }
  __syncthreads();

  if (tid == 0) {
    const int* tg = tags + (size_t)b * Tlen;
    const unsigned char* mk = mask + (size_t)b * Tlen;

    // gold path score
    int prev = tg[0];
    float score = start_t[prev] + em[0][prev];
    int cnt = mk[0] ? 1 : 0;
    for (int t = 1; t < Tlen; ++t) {
      int cur = tg[t];
      float m = mk[t] ? 1.0f : 0.0f;
      score += (trsh[prev * Kk + cur] + em[t][cur]) * m;
      cnt += mk[t] ? 1 : 0;
      prev = cur;
    }
    int last = cnt - 1; if (last < 0) last = 0;
    score += end_t[tg[last]];

    // forward algorithm (logZ)
    float al[Kk], na[Kk];
#pragma unroll
    for (int k = 0; k < Kk; ++k) al[k] = start_t[k] + em[0][k];
    for (int t = 1; t < Tlen; ++t) {
#pragma unroll
      for (int k2 = 0; k2 < Kk; ++k2) {
        float mx = -3.402823466e+38f;
#pragma unroll
        for (int k1 = 0; k1 < Kk; ++k1) {
          float v = al[k1] + trsh[k1 * Kk + k2];
          mx = fmaxf(mx, v);
        }
        float sm = 0.0f;
#pragma unroll
        for (int k1 = 0; k1 < Kk; ++k1)
          sm += expf(al[k1] + trsh[k1 * Kk + k2] - mx);
        na[k2] = mx + logf(sm) + em[t][k2];
      }
      if (mk[t]) {
#pragma unroll
        for (int k = 0; k < Kk; ++k) al[k] = na[k];
      }
    }
    float mz = -3.402823466e+38f;
#pragma unroll
    for (int k = 0; k < Kk; ++k) mz = fmaxf(mz, al[k] + end_t[k]);
    float sz = 0.0f;
#pragma unroll
    for (int k = 0; k < Kk; ++k) sz += expf(al[k] + end_t[k] - mz);
    float logZ = mz + logf(sz);

    nll_b[b] = logZ - score;   // mean over b gives -mean(score - logZ)
  }
}

__global__ void reduce_nll(const float* __restrict__ nll_b, float* __restrict__ out) {
  __shared__ float s[64];
  int t = threadIdx.x;
  s[t] = nll_b[t];
  __syncthreads();
  for (int o = 32; o > 0; o >>= 1) {
    if (t < o) s[t] += s[t + o];
    __syncthreads();
  }
  if (t == 0) out[0] = s[0] * (1.0f / 64.0f);
}

// ---------------- launch ----------------

extern "C" void kernel_launch(void* const* d_in, const int* in_sizes, int n_in,
                              void* d_out, int out_size, void* d_ws, size_t ws_size,
                              hipStream_t stream) {
  (void)in_sizes; (void)n_in; (void)out_size; (void)ws_size;
  const int* sent            = (const int*)d_in[0];
  const int* tags            = (const int*)d_in[1];
  const unsigned char* mask  = (const unsigned char*)d_in[2];
  const float* emb           = (const float*)d_in[3];
  const float* w_ih_f        = (const float*)d_in[4];
  const float* w_hh_f        = (const float*)d_in[5];
  const float* b_ih_f        = (const float*)d_in[6];
  const float* b_hh_f        = (const float*)d_in[7];
  const float* w_ih_b        = (const float*)d_in[8];
  const float* w_hh_b        = (const float*)d_in[9];
  const float* b_ih_b        = (const float*)d_in[10];
  const float* b_hh_b        = (const float*)d_in[11];
  const float* w_out         = (const float*)d_in[12];
  const float* b_out         = (const float*)d_in[13];
  const float* start_t       = (const float*)d_in[14];
  const float* end_t         = (const float*)d_in[15];
  const float* trans         = (const float*)d_in[16];
  float* out = (float*)d_out;

  char* ws = (char*)d_ws;
  size_t off = 0;
  auto carve = [&](size_t bytes) -> char* {
    char* p = ws + off;
    off += (bytes + 255) & ~(size_t)255;
    return p;
  };
  _Float16* x16 = (_Float16*)carve((size_t)Tlen * Bb * Ee * 2);   // 8.4 MB
  _Float16* WTf = (_Float16*)carve((size_t)KDIM * G4 * 2);        // 1 MB
  _Float16* WTb = (_Float16*)carve((size_t)KDIM * G4 * 2);        // 1 MB
  float*    bsf = (float*)carve((size_t)G4 * 4);
  float*    bsb = (float*)carve((size_t)G4 * 4);
  _Float16* hsf = (_Float16*)carve((size_t)Tlen * Bb * Hh * 2);   // 8.4 MB
  _Float16* hsb = (_Float16*)carve((size_t)Tlen * Bb * Hh * 2);   // 8.4 MB
  float*    nll = (float*)carve(64 * 4);

  embed_gather_f16<<<Tlen * Bb, Ee, 0, stream>>>(sent, emb, x16);
  build_wt<<<dim3(KDIM, 2), 256, 0, stream>>>(w_ih_f, w_hh_f, w_ih_b, w_hh_b, WTf, WTb);
  build_bsum<<<4, 256, 0, stream>>>(b_ih_f, b_hh_f, b_ih_b, b_hh_b, bsf, bsb);

  size_t shmem = (size_t)Bb * KDIM * 2 + (size_t)Bb * G4 * 2 + (size_t)Bb * Hh * 4; // 256 KB < 320 KB/WGP
  lstm_scan<<<2, 1024, shmem, stream>>>(x16, WTf, WTb, bsf, bsb, hsf, hsb);

  emis_crf<<<Bb, 256, 0, stream>>>(hsf, hsb, w_out, b_out, start_t, end_t, trans, tags, mask, nll);
  reduce_nll<<<1, 64, 0, stream>>>(nll, out);
}
